// GINetConvLayer_28381143892712
// MI455X (gfx1250) — matrix operand order repaired and checked by reading, not compiled, not run on hardware
//
#include <hip/hip_runtime.h>
#include <cstdint>

// Problem constants (reference: N,E,C,NEF,OUT,NUM_ANGLE = 20000,160000,32,1,128,7)
#define NN    20000
#define EE    160000
#define CC    32
#define OUTC  128
#define NANG  7
#define FF    65                 // 2*C + NEF
#define KK    (NANG * 2 * FF)    // 910
#define KP    928                // padded K = 29 * 32
#define KSTEPS (KP / 32)
#define LDSTR 936                // LDS A-tile row stride in halfs (928 + 8 pad)

typedef __attribute__((ext_vector_type(16))) _Float16 v16h;
typedef __attribute__((ext_vector_type(8)))  _Float16 v8h;
typedef __attribute__((ext_vector_type(4)))  _Float16 v4h;
typedef __attribute__((ext_vector_type(8)))  float    v8f;
typedef __attribute__((ext_vector_type(4)))  float    v4f;

// ---------------------------------------------------------------------------
// Stage 1: edge_f[e] = [ x[row[e]] | x[col[e]] | edge_attr[e] ]   [E, 65] f32
// ---------------------------------------------------------------------------
__global__ __launch_bounds__(256)
void build_edge_f_k(const float* __restrict__ x, const float* __restrict__ ea,
                    const int* __restrict__ row, const int* __restrict__ colv,
                    float* __restrict__ edge_f) {
    int idx = blockIdx.x * 256 + threadIdx.x;      // E*F = 10.4M
    if (idx >= EE * FF) return;
    int e = idx / FF;
    int f = idx - e * FF;
    float v;
    if (f < CC)            v = x[(size_t)row[e]  * CC + f];
    else if (f < 2 * CC)   v = x[(size_t)colv[e] * CC + (f - CC)];
    else                   v = ea[e];
    edge_f[idx] = v;
}

// ---------------------------------------------------------------------------
// Stage 1b: W (f32, [128,910]) -> Wh (f16, [128,928] zero-padded)
// ---------------------------------------------------------------------------
__global__ __launch_bounds__(256)
void convert_w_k(const float* __restrict__ W, _Float16* __restrict__ Wh) {
    int idx = blockIdx.x * 256 + threadIdx.x;      // 128*928 = 118784
    if (idx >= OUTC * KP) return;
    int o = idx / KP;
    int k = idx - o * KP;
    Wh[idx] = (k < KK) ? (_Float16)W[(size_t)o * KK + k] : (_Float16)0.0f;
}

// ---------------------------------------------------------------------------
// Stage 2: per line-graph pair, compute angle bin and scatter edge features
// into msg[E, 928] (fp32 atomics executed at L2). One wave32 per pair.
// ---------------------------------------------------------------------------
__global__ __launch_bounds__(256)
void pair_scatter_k(const int* __restrict__ lsrc, const int* __restrict__ ldst,
                    const int* __restrict__ row, const int* __restrict__ colv,
                    const float* __restrict__ pos, const float* __restrict__ edge_f,
                    float* __restrict__ msg, int L) {
    int w    = (blockIdx.x * 256 + threadIdx.x) >> 5;
    int lane = threadIdx.x & 31;
    if (w >= L) return;

    int a = lsrc[w];
    int b = ldst[w];
    int j = colv[a];           // shared node
    int i = row[a];
    int k = colv[b];

    float jx = pos[3 * j], jy = pos[3 * j + 1], jz = pos[3 * j + 2];
    float ux = pos[3 * i] - jx, uy = pos[3 * i + 1] - jy, uz = pos[3 * i + 2] - jz;
    float vx = pos[3 * k] - jx, vy = pos[3 * k + 1] - jy, vz = pos[3 * k + 2] - jz;

    float dot = ux * vx + uy * vy + uz * vz;
    float nu  = sqrtf(ux * ux + uy * uy + uz * uz);
    float nv  = sqrtf(vx * vx + vy * vy + vz * vz);
    float cs  = dot / (nu * nv + 1e-8f);
    cs = fminf(fmaxf(cs, -1.0f + 1e-7f), 1.0f - 1e-7f);
    float ang = acosf(cs);
    int abin = (int)(ang * ((float)NANG / 3.14159265358979323846f));
    abin = min(max(abin, 0), NANG - 1);

    float*       din  = msg + (size_t)b * KP + (abin * 2) * FF;       // incoming a->b
    float*       dout = msg + (size_t)a * KP + (abin * 2 + 1) * FF;   // outgoing b from a
    const float* fa   = edge_f + (size_t)a * FF;
    const float* fb   = edge_f + (size_t)b * FF;

    for (int f = lane; f < FF; f += 32) {
        atomicAdd(din  + f, fa[f]);   // global_atomic_add_f32
        atomicAdd(dout + f, fb[f]);
    }
}

// ---------------------------------------------------------------------------
// Stage 3: edge_hidden = msg @ W.T + b  via v_wmma_f32_16x16x32_f16.
// Block = 8 waves = one 16-edge tile x all 8 N-tiles.
//  - A tile (16 x 928) staged ONCE per block into LDS as f16 (30 KB of the
//    320 KB WGP LDS), shared by all 8 waves -> ds_load_b128 in the hot loop.
//  - B rows come from the 237 KB L2-resident Wh (two 16B loads/lane/step).
//  - Epilogue fuses bias + node scatter: out[col[e], :] += D[e, :].
// ---------------------------------------------------------------------------
__global__ __launch_bounds__(256)
void gemm_wmma_scatter_k(const float* __restrict__ msg, const _Float16* __restrict__ Wh,
                         const float* __restrict__ bmsg, const int* __restrict__ colv,
                         float* __restrict__ out) {
    __shared__ _Float16 lds_a[16 * LDSTR];

    const int e0 = blockIdx.x << 4;      // 16 edges per block

    // Cooperative stage: [16 x 928] f32 -> f16 into LDS (msg pad region is zeroed).
    for (int q = threadIdx.x; q < 16 * (KP / 4); q += 256) {
        int r  = q / (KP / 4);
        int k4 = (q - r * (KP / 4)) * 4;
        v4f a = *(const v4f*)(msg + (size_t)(e0 + r) * KP + k4);
        v4h h;
        h[0] = (_Float16)a[0];
        h[1] = (_Float16)a[1];
        h[2] = (_Float16)a[2];
        h[3] = (_Float16)a[3];
        *(v4h*)(&lds_a[r * LDSTR + k4]) = h;   // ds_store_b64
    }
    __syncthreads();

    const int wave = threadIdx.x >> 5;
    const int lane = threadIdx.x & 31;
    const int n0   = wave << 4;          // 16 output cols per wave
    const int nn   = lane & 15;
    const int hi   = lane >> 4;

    // A fragment source: lane nn = M row; hi selects K subgroup {0-7,16-23}/{8-15,24-31}
    const _Float16* arow = lds_a + nn * LDSTR + hi * 8;
    // B fragment source: lane nn = N col (Wh row n0+nn); hi selects K subgroup {0-7,16-23}/{8-15,24-31}
    const _Float16* brow = Wh + (size_t)(n0 + nn) * KP + hi * 8;

    const float bias = bmsg[n0 + nn];
    v8f c;
#pragma unroll
    for (int r = 0; r < 8; ++r) c[r] = bias;

    for (int ks = 0; ks < KSTEPS; ++ks) {
        const int kb = ks * 32;
        __builtin_prefetch(brow + kb + 64, 0, 1);   // global_prefetch_b8 (B stream)

        v8h a0 = *(const v8h*)(arow + kb);          // ds_load_b128: K = base+0..7
        v8h a1 = *(const v8h*)(arow + kb + 16);     // ds_load_b128: K = base+16..23
        v16h af;
#pragma unroll
        for (int j = 0; j < 8; ++j) { af[j] = a0[j]; af[8 + j] = a1[j]; }

        v8h b0 = *(const v8h*)(brow + kb);          // K = base+0..7
        v8h b1 = *(const v8h*)(brow + kb + 16);     // K = base+16..23
        v16h bf;
#pragma unroll
        for (int j = 0; j < 8; ++j) { bf[j] = b0[j]; bf[8 + j] = b1[j]; }

        c = __builtin_amdgcn_wmma_f32_16x16x32_f16(
                /*neg_a=*/false, af, /*neg_b=*/false, bf,
                /*c_mod=*/(short)0, c, /*reuse_a=*/false, /*reuse_b=*/false);
    }

    // Epilogue: D[M = hi*8 + r][N = nn] -> atomic scatter into destination nodes.
#pragma unroll
    for (int r = 0; r < 8; ++r) {
        int e = e0 + hi * 8 + r;
        atomicAdd(out + (size_t)colv[e] * OUTC + n0 + nn, c[r]);
    }
}

// ---------------------------------------------------------------------------
extern "C" void kernel_launch(void* const* d_in, const int* in_sizes, int n_in,
                              void* d_out, int out_size, void* d_ws, size_t ws_size,
                              hipStream_t stream) {
    const float* x    = (const float*)d_in[0];
    const float* ea   = (const float*)d_in[1];
    const float* pos  = (const float*)d_in[2];
    const float* W    = (const float*)d_in[3];
    const float* bm   = (const float*)d_in[4];
    const int*   eidx = (const int*)d_in[5];   // [2, E]
    const int*   lsrc = (const int*)d_in[6];
    const int*   ldst = (const int*)d_in[7];
    const int    L    = in_sizes[6];

    const int* row  = eidx;
    const int* colv = eidx + EE;

    // Workspace carve-up
    char*  ws  = (char*)d_ws;
    size_t off = 0;
    float* msg = (float*)(ws + off);     off += (size_t)EE * KP * sizeof(float);   // 594 MB
    float* edge_f = (float*)(ws + off);  off += (size_t)EE * FF * sizeof(float);   // 41.6 MB
    off = (off + 255) & ~(size_t)255;
    _Float16* Wh = (_Float16*)(ws + off); off += (size_t)OUTC * KP * sizeof(_Float16);

    // Zero accumulators each call (graph-capture-safe memsets)
    hipMemsetAsync(msg, 0, (size_t)EE * KP * sizeof(float), stream);
    hipMemsetAsync(d_out, 0, (size_t)out_size * sizeof(float), stream);

    build_edge_f_k<<<(EE * FF + 255) / 256, 256, 0, stream>>>(x, ea, row, colv, edge_f);
    convert_w_k<<<(OUTC * KP + 255) / 256, 256, 0, stream>>>(W, Wh);
    pair_scatter_k<<<(L + 7) / 8, 256, 0, stream>>>(lsrc, ldst, row, colv, pos, edge_f, msg, L);
    gemm_wmma_scatter_k<<<EE / 16, 256, 0, stream>>>(msg, Wh, bm, colv, (float*)d_out);
}
